// MetalAttention_678604833221
// MI455X (gfx1250) — matrix operand order repaired and checked by reading, not compiled, not run on hardware
//
#include <hip/hip_runtime.h>
#include <cstddef>
#include <cstdint>

typedef __bf16 bf16_t;
typedef __attribute__((ext_vector_type(16))) __bf16 v16bf;
typedef __attribute__((ext_vector_type(8)))  __bf16 v8bf;
typedef __attribute__((ext_vector_type(8)))  float  v8f;

#define S_LEN 2048
#define NH    32
#define NKVH  8
#define HD    128
#define HIDS  4096
#define KVS   1024
#define ATTN_SCALE 0.08838834764831845f  // 1/sqrt(128)

// ---------------- helpers ----------------

__device__ __forceinline__ v16bf pack16(v8bf lo, v8bf hi) {
  v16bf r;
#pragma unroll
  for (int i = 0; i < 8; ++i) { r[i] = lo[i]; r[i + 8] = hi[i]; }
  return r;
}

__device__ __forceinline__ v8f zero8() {
  v8f z = {0.f, 0.f, 0.f, 0.f, 0.f, 0.f, 0.f, 0.f};
  return z;
}

__device__ __forceinline__ v8f wmma_bf16(v16bf a, v16bf b, v8f c) {
  // D = A(16x32 bf16) * B(32x16 bf16) + C(16x16 f32)
  return __builtin_amdgcn_wmma_f32_16x16x32_bf16(false, a, false, b, (short)0, c,
                                                 false, false);
}

// reduce across the 16 lanes that hold the same output row (lanes 0-15 / 16-31)
__device__ __forceinline__ float redmax16(float v) {
  v = fmaxf(v, __shfl_xor(v, 1, 32));
  v = fmaxf(v, __shfl_xor(v, 2, 32));
  v = fmaxf(v, __shfl_xor(v, 4, 32));
  v = fmaxf(v, __shfl_xor(v, 8, 32));
  return v;
}
__device__ __forceinline__ float redsum16(float v) {
  v += __shfl_xor(v, 1, 32);
  v += __shfl_xor(v, 2, 32);
  v += __shfl_xor(v, 4, 32);
  v += __shfl_xor(v, 8, 32);
  return v;
}

// ---------------- elementwise kernels ----------------

__global__ void f32_to_bf16_kernel(const float* __restrict__ in,
                                   bf16_t* __restrict__ out, int n) {
  int idx = blockIdx.x * blockDim.x + threadIdx.x;
  if (idx < n) out[idx] = (bf16_t)in[idx];
}

// rotate-half RoPE; in fp32 [S][nh][128], out bf16 same layout
__global__ void rope_bf16_kernel(const float* __restrict__ in,
                                 bf16_t* __restrict__ out, int nh) {
  int idx = blockIdx.x * blockDim.x + threadIdx.x;
  int total = S_LEN * nh * 64;
  if (idx >= total) return;
  int i = idx & 63;
  int rest = idx >> 6;
  int hh = rest % nh;
  int s = rest / nh;
  float inv = powf(10000.0f, -(float)i * (1.0f / 64.0f));
  float ang = (float)s * inv;
  float sn, cs;
  __sincosf(ang, &sn, &cs);
  size_t base = ((size_t)s * nh + hh) * HD;
  float x1 = in[base + i];
  float x2 = in[base + 64 + i];
  out[base + i]      = (bf16_t)(x1 * cs - x2 * sn);
  out[base + 64 + i] = (bf16_t)(x2 * cs + x1 * sn);
}

// ---- fused-dequant WMMA GEMM: C[M,N]f32 = A[M,K]bf16 * dequant(codes,scales) ----
// codes: [K,N] int32 in [0,16); scales: [K/128, N] f32; w = (code-8)*0.5*scale.
// Block tile 128(M) x 64(N), BK=32. 4 waves; wave w owns rows [32w, 32w+32).
// Per K-step per wave: 2 A-frags, 4 B-frags, 8 WMMAs.
__global__ __launch_bounds__(128) void gemm_bf16_q4_kernel(
    const bf16_t* __restrict__ A, const int* __restrict__ codes,
    const float* __restrict__ scales, float* __restrict__ C,
    int M, int N, int K) {
  __shared__ bf16_t Bt[64][32];  // transposed dequantized B tile: Bt[n][k]
  const int tid = threadIdx.x;
  const int w = tid >> 5, lane = tid & 31;
  const int hl = lane >> 4, l15 = lane & 15;
  const int m0 = blockIdx.y * 128 + 32 * w;
  const int nb0 = blockIdx.x * 64;
  const int kkb = tid >> 3;        // 0..15
  const int nc0 = (tid & 7) * 8;   // 0..56

  v8f acc[2][4];
#pragma unroll
  for (int mt = 0; mt < 2; ++mt)
#pragma unroll
    for (int nt = 0; nt < 4; ++nt) acc[mt][nt] = zero8();

  float sc[8];  // 0.5 * scale for this thread's 8 columns; reload per 128-K group

  for (int k0 = 0; k0 < K; k0 += 32) {
    // K-tiles of 32 never straddle a 128-group boundary -> group fixed per tile
    if ((k0 & 127) == 0) {
      const float* sp = scales + (size_t)(k0 >> 7) * N + nb0 + nc0;
#pragma unroll
      for (int j = 0; j < 8; ++j) sc[j] = 0.5f * sp[j];
    }
    __syncthreads();
    // cooperative dequant of codes[k0..k0+31][nb0..nb0+63] into Bt (transposed)
#pragma unroll
    for (int pass = 0; pass < 2; ++pass) {
      int kk = kkb + 16 * pass;
      const int* cp = codes + (size_t)(k0 + kk) * N + nb0 + nc0;
      if (k0 + 32 < K) __builtin_prefetch(cp + (size_t)32 * N, 0, 1);
      int4 c0 = *(const int4*)cp;
      int4 c1 = *(const int4*)(cp + 4);
      Bt[nc0 + 0][kk] = (bf16_t)((float)(c0.x - 8) * sc[0]);
      Bt[nc0 + 1][kk] = (bf16_t)((float)(c0.y - 8) * sc[1]);
      Bt[nc0 + 2][kk] = (bf16_t)((float)(c0.z - 8) * sc[2]);
      Bt[nc0 + 3][kk] = (bf16_t)((float)(c0.w - 8) * sc[3]);
      Bt[nc0 + 4][kk] = (bf16_t)((float)(c1.x - 8) * sc[4]);
      Bt[nc0 + 5][kk] = (bf16_t)((float)(c1.y - 8) * sc[5]);
      Bt[nc0 + 6][kk] = (bf16_t)((float)(c1.z - 8) * sc[6]);
      Bt[nc0 + 7][kk] = (bf16_t)((float)(c1.w - 8) * sc[7]);
    }
    __syncthreads();
    // A fragments (16x32): lane row = m0 + 16*mt + l15; chunks at 8*hl, 16+8*hl
    v16bf af[2];
#pragma unroll
    for (int mt = 0; mt < 2; ++mt) {
      const bf16_t* ap = A + (size_t)(m0 + 16 * mt + l15) * K + k0 + 8 * hl;
      af[mt] = pack16(*(const v8bf*)ap, *(const v8bf*)(ap + 16));
    }
#pragma unroll
    for (int nt = 0; nt < 4; ++nt) {
      // B fragment: lane column n = 16*nt + l15; chunks at 8*hl, 16+8*hl
      const bf16_t* bp = &Bt[16 * nt + l15][8 * hl];
      v16bf bfr = pack16(*(const v8bf*)bp, *(const v8bf*)(bp + 16));
      acc[0][nt] = wmma_bf16(af[0], bfr, acc[0][nt]);
      acc[1][nt] = wmma_bf16(af[1], bfr, acc[1][nt]);
    }
  }
#pragma unroll
  for (int mt = 0; mt < 2; ++mt)
#pragma unroll
    for (int nt = 0; nt < 4; ++nt)
#pragma unroll
      for (int r = 0; r < 8; ++r)
        C[(size_t)(m0 + 16 * mt + r + 8 * hl) * N + nb0 + 16 * nt + l15] =
            acc[mt][nt][r];
}

// ---------------- flash attention (causal, GQA 4:1), all-WMMA ----------------
// grid: (S/64, NH); 4 waves; wave w handles queries [qb0+16w, qb0+16w+16)
__global__ __launch_bounds__(128) void attn_kernel(
    const bf16_t* __restrict__ qb, const bf16_t* __restrict__ kb,
    const bf16_t* __restrict__ vb, bf16_t* __restrict__ ob) {
  __shared__ bf16_t Vt[HD][32];       // V tile transposed: Vt[d][k_local]
  __shared__ bf16_t Pl[4][16][32];    // per-wave P staging (C-layout -> A-layout)
  const int tid = threadIdx.x;
  const int w = tid >> 5, lane = tid & 31;
  const int hl = lane >> 4, l15 = lane & 15;
  const int h = blockIdx.y;
  const int kvh = h >> 2;             // GQA: 4 query heads per kv head
  const int qb0 = blockIdx.x * 64;
  const int qbase = qb0 + 16 * w;

  // Q fragments: 4 chunks of 16x32 over D=128
  v16bf qf[4];
  {
    const bf16_t* qp = qb + ((size_t)(qbase + l15) * NH + h) * HD;
#pragma unroll
    for (int c = 0; c < 4; ++c) {
      v8bf lo = *(const v8bf*)(qp + 32 * c + 8 * hl);
      v8bf hi = *(const v8bf*)(qp + 32 * c + 16 + 8 * hl);
      qf[c] = pack16(lo, hi);
    }
  }

  v8f Of[8];
  float mrow[8], lrow[8];
#pragma unroll
  for (int t = 0; t < 8; ++t) Of[t] = zero8();
#pragma unroll
  for (int r = 0; r < 8; ++r) { mrow[r] = -3.0e38f; lrow[r] = 0.0f; }

  const int nkb = qb0 / 32 + 2;  // key blocks of 32 needed by this workgroup
  for (int kbI = 0; kbI < nkb; ++kbI) {
    const int kstart = kbI * 32;
    __syncthreads();
    // cooperative transposed load of V block [32 keys x 128 d] into Vt
    {
      int kk = tid & 31;
      int d0 = (tid >> 5) * 32;
      const bf16_t* vp = vb + ((size_t)(kstart + kk) * NKVH + kvh) * HD + d0;
#pragma unroll
      for (int jj = 0; jj < 4; ++jj) {
        v8bf vv = *(const v8bf*)(vp + 8 * jj);
#pragma unroll
        for (int i = 0; i < 8; ++i) Vt[d0 + 8 * jj + i][kk] = vv[i];
      }
    }
    __syncthreads();

    // scores S[16q x 32k] as two 16x16 tiles
    v8f Sf[2];
    Sf[0] = zero8(); Sf[1] = zero8();
#pragma unroll
    for (int kt = 0; kt < 2; ++kt) {
      const bf16_t* kp =
          kb + ((size_t)(kstart + 16 * kt + l15) * NKVH + kvh) * HD + 8 * hl;
#pragma unroll
      for (int c = 0; c < 4; ++c) {
        v16bf kf = pack16(*(const v8bf*)(kp + 32 * c), *(const v8bf*)(kp + 32 * c + 16));
        Sf[kt] = wmma_bf16(qf[c], kf, Sf[kt]);
      }
    }
    // scale + causal mask (column key = kstart+16kt+l15; row q = qbase+r+8*hl)
#pragma unroll
    for (int kt = 0; kt < 2; ++kt) {
      int key = kstart + 16 * kt + l15;
#pragma unroll
      for (int r = 0; r < 8; ++r) {
        int qrow = qbase + r + 8 * hl;
        float s = Sf[kt][r] * ATTN_SCALE;
        Sf[kt][r] = (key > qrow) ? -1.0e30f : s;
      }
    }
    // online softmax (row stats replicated across the 16 lanes of each row)
    float alpha[8];
#pragma unroll
    for (int r = 0; r < 8; ++r) {
      float t = fmaxf(Sf[0][r], Sf[1][r]);
      t = redmax16(t);
      float mn = fmaxf(mrow[r], t);
      alpha[r] = __expf(mrow[r] - mn);
      mrow[r] = mn;
    }
#pragma unroll
    for (int kt = 0; kt < 2; ++kt)
#pragma unroll
      for (int r = 0; r < 8; ++r) Sf[kt][r] = __expf(Sf[kt][r] - mrow[r]);
#pragma unroll
    for (int r = 0; r < 8; ++r) {
      float ps = redsum16(Sf[0][r] + Sf[1][r]);
      lrow[r] = lrow[r] * alpha[r] + ps;
    }
    // stage P (C-layout) into LDS, reload in A-layout
#pragma unroll
    for (int kt = 0; kt < 2; ++kt)
#pragma unroll
      for (int r = 0; r < 8; ++r)
        Pl[w][r + 8 * hl][16 * kt + l15] = (bf16_t)Sf[kt][r];
    asm volatile("s_wait_dscnt 0" ::: "memory");  // same-wave LDS RAW fence
    const bf16_t* pp = &Pl[w][l15][8 * hl];
    v16bf pf = pack16(*(const v8bf*)pp, *(const v8bf*)(pp + 16));
    // O = diag(alpha)*O + P*V
#pragma unroll
    for (int t = 0; t < 8; ++t) {
#pragma unroll
      for (int r = 0; r < 8; ++r) Of[t][r] *= alpha[r];
      const bf16_t* vtp = &Vt[16 * t + l15][8 * hl];
      v16bf vf = pack16(*(const v8bf*)vtp, *(const v8bf*)(vtp + 16));
      Of[t] = wmma_bf16(pf, vf, Of[t]);
    }
  }
  // epilogue: normalize and store bf16 [S][H*D]
#pragma unroll
  for (int t = 0; t < 8; ++t)
#pragma unroll
    for (int r = 0; r < 8; ++r) {
      float o = Of[t][r] / lrow[r];
      ob[(size_t)(qbase + r + 8 * hl) * HIDS + h * HD + 16 * t + l15] = (bf16_t)o;
    }
}

// ---------------- host launcher ----------------

extern "C" void kernel_launch(void* const* d_in, const int* in_sizes, int n_in,
                              void* d_out, int out_size, void* d_ws, size_t ws_size,
                              hipStream_t stream) {
  (void)in_sizes; (void)n_in; (void)out_size; (void)ws_size;
  const float* x        = (const float*)d_in[0];
  const int*   q_codes  = (const int*)d_in[1];
  const int*   k_codes  = (const int*)d_in[2];
  const int*   v_codes  = (const int*)d_in[3];
  const int*   o_codes  = (const int*)d_in[4];
  const float* q_scales = (const float*)d_in[5];
  const float* k_scales = (const float*)d_in[6];
  const float* v_scales = (const float*)d_in[7];
  const float* o_scales = (const float*)d_in[8];
  float* out = (float*)d_out;

  char* ws = (char*)d_ws;
  size_t off = 0;
  auto carve = [&](size_t bytes) -> void* {
    void* p = ws + off;
    off += (bytes + 255) & ~(size_t)255;
    return p;
  };
  bf16_t* xb  = (bf16_t*)carve((size_t)S_LEN * HIDS * 2);
  float*  qf  = (float*)carve((size_t)S_LEN * HIDS * 4);
  float*  kf  = (float*)carve((size_t)S_LEN * KVS * 4);
  float*  vf  = (float*)carve((size_t)S_LEN * KVS * 4);
  bf16_t* qbf = (bf16_t*)carve((size_t)S_LEN * HIDS * 2);
  bf16_t* kbf = (bf16_t*)carve((size_t)S_LEN * KVS * 2);
  bf16_t* vbf = (bf16_t*)carve((size_t)S_LEN * KVS * 2);
  bf16_t* abf = (bf16_t*)qf;  // alias: q_f32 dead after RoPE

  const int thr = 256;
  f32_to_bf16_kernel<<<(S_LEN * HIDS + thr - 1) / thr, thr, 0, stream>>>(
      x, xb, S_LEN * HIDS);

  gemm_bf16_q4_kernel<<<dim3(HIDS / 64, S_LEN / 128), 128, 0, stream>>>(
      xb, q_codes, q_scales, qf, S_LEN, HIDS, HIDS);
  gemm_bf16_q4_kernel<<<dim3(KVS / 64, S_LEN / 128), 128, 0, stream>>>(
      xb, k_codes, k_scales, kf, S_LEN, KVS, HIDS);
  gemm_bf16_q4_kernel<<<dim3(KVS / 64, S_LEN / 128), 128, 0, stream>>>(
      xb, v_codes, v_scales, vf, S_LEN, KVS, HIDS);

  rope_bf16_kernel<<<(S_LEN * NH * 64 + thr - 1) / thr, thr, 0, stream>>>(qf, qbf, NH);
  rope_bf16_kernel<<<(S_LEN * NKVH * 64 + thr - 1) / thr, thr, 0, stream>>>(kf, kbf, NKVH);
  f32_to_bf16_kernel<<<(S_LEN * KVS + thr - 1) / thr, thr, 0, stream>>>(
      vf, vbf, S_LEN * KVS);

  attn_kernel<<<dim3(S_LEN / 64, NH), 128, 0, stream>>>(qbf, kbf, vbf, abf);

  gemm_bf16_q4_kernel<<<dim3(HIDS / 64, S_LEN / 128), 128, 0, stream>>>(
      abf, o_codes, o_scales, out, S_LEN, HIDS, HIDS);
}
